// MultiHeadAttention_2388001816593
// MI455X (gfx1250) — matrix operand order repaired and checked by reading, not compiled, not run on hardware
//
#include <hip/hip_runtime.h>
#include <hip/hip_bf16.h>
#include <stdint.h>

#define D_MODEL 1024
#define N_HEADS 16
#define HEAD_DIM 64
#define BB 4
#define SS 1024

typedef __attribute__((ext_vector_type(16))) __bf16 v16bf_t;
typedef __attribute__((ext_vector_type(8)))  float  v8f_t;
typedef unsigned int v4u_t __attribute__((ext_vector_type(4)));
typedef int          v8i_t __attribute__((ext_vector_type(8)));
typedef int          v4i_t __attribute__((ext_vector_type(4)));

union BfVec  { uint4 u[2]; v16bf_t v; };
union Pack8  { unsigned short s[8]; uint4 u; };
union Pack4  { unsigned short s[4]; uint2 u; };

#if defined(__AMDGCN__) && __has_builtin(__builtin_amdgcn_tensor_load_to_lds) && \
    __has_builtin(__builtin_amdgcn_s_wait_tensorcnt)
#define USE_TDM 1
#else
#define USE_TDM 0
#endif

static __device__ __forceinline__ unsigned short f2bf(float f) {
  unsigned int u = __float_as_uint(f);
  unsigned int r = (u + 0x7FFFu + ((u >> 16) & 1u)) >> 16;   // round-to-nearest-even
  return (unsigned short)r;
}

static __device__ __forceinline__ v8f_t wmma_bf16(v16bf_t a, v16bf_t b, v8f_t c) {
  return __builtin_amdgcn_wmma_f32_16x16x32_bf16(false, a, false, b, (short)0, c, false, false);
}

static __device__ __forceinline__ uint32_t lds_off(const void* p) {
  // flat shared pointer: low 32 bits are the LDS byte offset (ISA 10.2 aperture rules)
  return (uint32_t)(uintptr_t)p;
}

#if USE_TDM
// 2D bf16 tile DMA: tile_x elems wide, tile_rows rows, row stride in elems.
// D# packing per CDNA5 ISA 8.3/8.4 (data_size=1 => 2 bytes, type=2 "image").
static __device__ __forceinline__ void tdm_load_2d_bf16(
    const unsigned short* gsrc, uint32_t ldsoff,
    int tile_x, int tile_rows, int row_stride) {
  const unsigned long long ga = (unsigned long long)(uintptr_t)gsrc;
  const int td0 = row_stride;        // tensor dim0 (elems)
  const int td1 = 1 << 20;           // generous row bound
  v4u_t g0 = { 1u,                                   // count=1, user descriptor
               ldsoff,                               // lds_addr (bytes)
               (unsigned int)ga,                     // global_addr[31:0]
               (unsigned int)(ga >> 32) | (2u << 30) // global_addr[56:32] | type=2
             };
  v8i_t g1 = { (int)(1u << 16),                                    // data_size=1 (2B)
               (int)((td0 & 0xFFFF) << 16),                        // tensor_dim0 lo @bit48
               (int)(((td0 >> 16) & 0xFFFF) | ((td1 & 0xFFFF) << 16)),
               (int)(((td1 >> 16) & 0xFFFF) | ((tile_x & 0xFFFF) << 16)), // tile_dim0
               (int)(tile_rows & 0xFFFF),                          // tile_dim1 (tile_dim2=0)
               row_stride,                                         // tensor_dim0_stride lo
               0, 0 };
  const v4i_t z4 = {0, 0, 0, 0};
#if __clang_major__ >= 23
  const v8i_t z8 = {0, 0, 0, 0, 0, 0, 0, 0};
  __builtin_amdgcn_tensor_load_to_lds(g0, g1, z4, z4, z8, 0);
#else
  __builtin_amdgcn_tensor_load_to_lds(g0, g1, z4, z4, 0);
#endif
}
#define TDM_WAIT() __builtin_amdgcn_s_wait_tensorcnt(0)
#endif

// LDS tile readers in WMMA operand layouts (ISA 7.12.2, wave32).
static __device__ __forceinline__ v16bf_t lds_a_tile(const unsigned short* tile, int rs) {
  const int lane = threadIdx.x & 31;
  const unsigned short* p = tile + (lane & 15) * rs + (lane >> 4) * 8;
  BfVec bv;
  bv.u[0] = *(const uint4*)p;          // K = c*8 .. c*8+7
  bv.u[1] = *(const uint4*)(p + 16);   // K = c*8+16 .. c*8+23
  return bv.v;
}
static __device__ __forceinline__ v16bf_t lds_b_tile(const unsigned short* tile,
                                                     int n0, int rs, int k0) {
  const int lane = threadIdx.x & 31;
  const unsigned short* p = tile + (n0 + (lane & 15)) * rs + k0 + ((lane >> 4) << 3);
  BfVec bv;
  bv.u[0] = *(const uint4*)p;
  bv.u[1] = *(const uint4*)(p + 16);
  return bv.v;
}

// ---- f32 -> bf16 conversion (x4 vectorized) ------------------------------
__global__ __launch_bounds__(256) void cvt_kernel(const float* __restrict__ in,
                                                  unsigned short* __restrict__ out, int n4) {
  int i = blockIdx.x * blockDim.x + threadIdx.x;
  if (i < n4) {
    float4 f = ((const float4*)in)[i];
    Pack4 pk;
    pk.s[0] = f2bf(f.x); pk.s[1] = f2bf(f.y); pk.s[2] = f2bf(f.z); pk.s[3] = f2bf(f.w);
    ((uint2*)out)[i] = pk.u;
  }
}

// ---- GEMM: Y = X(M x K) * W(N x K)^T + bias ------------------------------
// Block: 128 rows (8 waves x 16) x 128 cols. W tile TDM-staged once per block
// (shared by all 8 waves), X tiles TDM-staged per wave. Double buffered.
// Inner loop: batch all LDS operand loads (one dscnt wait), then 8 WMMAs.
__global__ __launch_bounds__(256) void gemm_bf16_kernel(
    const unsigned short* __restrict__ X,
    const unsigned short* __restrict__ W,
    const float* __restrict__ bias,
    unsigned short* __restrict__ Ybf,
    float* __restrict__ Yf,
    unsigned short* __restrict__ Yt,
    int Kdim, int Ncols) {
  __shared__ __align__(16) unsigned short Wt[2][128 * 32];    // 8 KB x2
  __shared__ __align__(16) unsigned short At[2][8][16 * 32];  // 1 KB x8 x2

  const int lane = threadIdx.x & 31;
  const int wave = threadIdx.x >> 5;
  const int mb = (blockIdx.y * 8 + wave) * 16;
  const int nb = blockIdx.x * 128;
  const int hl = lane >> 4, nloc = lane & 15;

  v8f_t acc[8];
  const v8f_t zero = {0.f,0.f,0.f,0.f,0.f,0.f,0.f,0.f};
#pragma unroll
  for (int j = 0; j < 8; ++j) acc[j] = zero;

#if USE_TDM
  if (wave == 0)
    tdm_load_2d_bf16(W + (size_t)nb * Kdim, lds_off(&Wt[0][0]), 32, 128, Kdim);
  tdm_load_2d_bf16(X + (size_t)mb * Kdim, lds_off(&At[0][wave][0]), 32, 16, Kdim);
  TDM_WAIT();
#else
  {
    const int t = threadIdx.x;
#pragma unroll
    for (int i = 0; i < 2; ++i) {                     // 128x32 W tile: 512 uint4
      int idx = t + i * 256, row = idx >> 2, ch = idx & 3;
      *(uint4*)&Wt[0][row * 32 + ch * 8] =
          *(const uint4*)&W[(size_t)(nb + row) * Kdim + ch * 8];
    }
#pragma unroll
    for (int i = 0; i < 2; ++i) {                     // 16x32 A tile: 64 uint4
      int idx = lane + i * 32, row = idx >> 2, ch = idx & 3;
      *(uint4*)&At[0][wave][row * 32 + ch * 8] =
          *(const uint4*)&X[(size_t)(mb + row) * Kdim + ch * 8];
    }
  }
#endif
  __syncthreads();

  for (int kc = 0; kc < Kdim; kc += 32) {
    const int cur = (kc >> 5) & 1, nxt = cur ^ 1;
    if (kc + 32 < Kdim) {
#if USE_TDM
      if (wave == 0)
        tdm_load_2d_bf16(W + (size_t)nb * Kdim + kc + 32, lds_off(&Wt[nxt][0]), 32, 128, Kdim);
      tdm_load_2d_bf16(X + (size_t)mb * Kdim + kc + 32, lds_off(&At[nxt][wave][0]), 32, 16, Kdim);
#else
      const int t = threadIdx.x;
#pragma unroll
      for (int i = 0; i < 2; ++i) {
        int idx = t + i * 256, row = idx >> 2, ch = idx & 3;
        *(uint4*)&Wt[nxt][row * 32 + ch * 8] =
            *(const uint4*)&W[(size_t)(nb + row) * Kdim + kc + 32 + ch * 8];
      }
#pragma unroll
      for (int i = 0; i < 2; ++i) {
        int idx = lane + i * 32, row = idx >> 2, ch = idx & 3;
        *(uint4*)&At[nxt][wave][row * 32 + ch * 8] =
            *(const uint4*)&X[(size_t)(mb + row) * Kdim + kc + 32 + ch * 8];
      }
#endif
    }
    // batch all LDS operand loads, then run the WMMA chain back-to-back
    const v16bf_t a = lds_a_tile(&At[cur][wave][0], 32);
    v16bf_t bt[8];
#pragma unroll
    for (int j = 0; j < 8; ++j) bt[j] = lds_b_tile(&Wt[cur][0], j * 16, 32, 0);
#pragma unroll
    for (int j = 0; j < 8; ++j) acc[j] = wmma_bf16(a, bt[j], acc[j]);
#if USE_TDM
    TDM_WAIT();
#endif
    __syncthreads();
  }

#pragma unroll
  for (int j = 0; j < 8; ++j) {
    const int n = nb + j * 16 + nloc;
    const float bsv = bias[n];
#pragma unroll
    for (int r = 0; r < 8; ++r) acc[j][r] += bsv;

    if (Ybf) {
#pragma unroll
      for (int r = 0; r < 8; ++r)
        Ybf[(size_t)(mb + 8 * hl + r) * Ncols + n] = f2bf(acc[j][r]);
    }
    if (Yf) {
#pragma unroll
      for (int r = 0; r < 8; ++r)
        Yf[(size_t)(mb + 8 * hl + r) * Ncols + n] = acc[j][r];
    }
    if (Yt) {                                        // transposed (B, N, SS) bf16
      Pack8 pk;
#pragma unroll
      for (int r = 0; r < 8; ++r) pk.s[r] = f2bf(acc[j][r]);
      const int bidx = mb / SS;
      const int srow = (mb - bidx * SS) + 8 * hl;
      *(uint4*)&Yt[((size_t)bidx * Ncols + n) * SS + srow] = pk.u;
    }
  }
}

// ---- Flash attention: one wave = 16 queries, 4 waves/WG ------------------
// K / Vt key-blocks are shared by all waves of the WG -> TDM-staged in LDS,
// double buffered; P goes through a per-wave LDS tile for the C->A transpose.
__global__ __launch_bounds__(128) void attn_kernel(
    const unsigned short* __restrict__ Q,    // (B,S,D) bf16
    const unsigned short* __restrict__ Km,   // (B,S,D) bf16
    const unsigned short* __restrict__ Vt,   // (B,D,S) bf16
    const unsigned char*  __restrict__ mask, // (B,1,S,S)
    unsigned short* __restrict__ Ctx) {      // (B,S,D) bf16
  __shared__ __align__(16) unsigned short kbuf[2][32 * 64];   // keys x dh, 4 KB x2
  __shared__ __align__(16) unsigned short vbuf[2][64 * 32];   // dh x keys, 4 KB x2
  __shared__ __align__(16) unsigned short plds[4][16][40];

  const int lane = threadIdx.x & 31;
  const int wave = threadIdx.x >> 5;
  const int b = blockIdx.z, h = blockIdx.y;
  const int qb = (blockIdx.x * 4 + wave) * 16;
  const int hl = lane >> 4, nloc = lane & 15;
  const int dh0 = h * HEAD_DIM;
  const size_t aOff = (size_t)b * SS * D_MODEL;
  const unsigned short* kbase = Km + aOff;
  const unsigned short* vbase = Vt + (size_t)b * D_MODEL * SS;

  const v8f_t zero = {0.f,0.f,0.f,0.f,0.f,0.f,0.f,0.f};
  v8f_t o[4];
#pragma unroll
  for (int j = 0; j < 4; ++j) o[j] = zero;
  float mrow[8], lrow[8];
#pragma unroll
  for (int r = 0; r < 8; ++r) { mrow[r] = -3.0e38f; lrow[r] = 0.f; }

  // Q operand tiles (held in registers for the whole loop)
  const v16bf_t qa0 = lds_a_tile(Q + aOff + (size_t)qb * D_MODEL + dh0, D_MODEL);
  const v16bf_t qa1 = lds_a_tile(Q + aOff + (size_t)qb * D_MODEL + dh0 + 32, D_MODEL);

  const unsigned char* mp = mask + (size_t)b * SS * SS;
  const float SCL = 0.125f * 1.44269504089f;   // log2(e)/sqrt(HEAD_DIM)

#if USE_TDM
  if (wave == 0) {
    tdm_load_2d_bf16(kbase + dh0, lds_off(&kbuf[0][0]), 64, 32, D_MODEL);
    tdm_load_2d_bf16(vbase + (size_t)dh0 * SS, lds_off(&vbuf[0][0]), 32, 64, SS);
    TDM_WAIT();
  }
#else
  {
    const int t = threadIdx.x;
#pragma unroll
    for (int i = 0; i < 2; ++i) {                    // 32x64: 256 uint4
      int idx = t + i * 128, row = idx >> 3, ch = idx & 7;
      *(uint4*)&kbuf[0][row * 64 + ch * 8] =
          *(const uint4*)&kbase[(size_t)row * D_MODEL + dh0 + ch * 8];
    }
#pragma unroll
    for (int i = 0; i < 2; ++i) {                    // 64x32: 256 uint4
      int idx = t + i * 128, row = idx >> 2, ch = idx & 3;
      *(uint4*)&vbuf[0][row * 32 + ch * 8] =
          *(const uint4*)&vbase[(size_t)(dh0 + row) * SS + ch * 8];
    }
  }
#endif
  __syncthreads();

  for (int kb = 0; kb < SS; kb += 32) {
    const int cur = (kb >> 5) & 1, nxt = cur ^ 1;
    if (kb + 32 < SS) {
#if USE_TDM
      if (wave == 0) {
        tdm_load_2d_bf16(kbase + (size_t)(kb + 32) * D_MODEL + dh0,
                         lds_off(&kbuf[nxt][0]), 64, 32, D_MODEL);
        tdm_load_2d_bf16(vbase + (size_t)dh0 * SS + kb + 32,
                         lds_off(&vbuf[nxt][0]), 32, 64, SS);
      }
#else
      const int t = threadIdx.x;
#pragma unroll
      for (int i = 0; i < 2; ++i) {
        int idx = t + i * 128, row = idx >> 3, ch = idx & 7;
        *(uint4*)&kbuf[nxt][row * 64 + ch * 8] =
            *(const uint4*)&kbase[(size_t)(kb + 32 + row) * D_MODEL + dh0 + ch * 8];
      }
#pragma unroll
      for (int i = 0; i < 2; ++i) {
        int idx = t + i * 128, row = idx >> 2, ch = idx & 3;
        *(uint4*)&vbuf[nxt][row * 32 + ch * 8] =
            *(const uint4*)&vbase[(size_t)(dh0 + row) * SS + kb + 32 + ch * 8];
      }
#endif
    }

    // ---- scores S = Q K^T : batch 4 LDS B-tiles, then 4 WMMAs
    v16bf_t kt[4];
    kt[0] = lds_b_tile(&kbuf[cur][0], 0, 64, 0);
    kt[1] = lds_b_tile(&kbuf[cur][0], 16, 64, 0);
    kt[2] = lds_b_tile(&kbuf[cur][0], 0, 64, 32);
    kt[3] = lds_b_tile(&kbuf[cur][0], 16, 64, 32);
    v8f_t s0 = zero, s1 = zero;
    s0 = wmma_bf16(qa0, kt[0], s0);
    s1 = wmma_bf16(qa0, kt[1], s1);
    s0 = wmma_bf16(qa1, kt[2], s0);
    s1 = wmma_bf16(qa1, kt[3], s1);

    // ---- mask (C layout: element (M=r+8*hl, N=nloc))
#pragma unroll
    for (int r = 0; r < 8; ++r) {
      const size_t qrow = (size_t)(qb + 8 * hl + r) * SS;
      s0[r] = mp[qrow + kb + nloc]      ? s0[r] : -3.0e38f;
      s1[r] = mp[qrow + kb + 16 + nloc] ? s1[r] : -3.0e38f;
    }

    // ---- online softmax (per output row, reduced over the 16-lane half)
#pragma unroll
    for (int r = 0; r < 8; ++r) {
      float v = fmaxf(s0[r], s1[r]);
      v = fmaxf(v, __shfl_xor(v, 1, 32));
      v = fmaxf(v, __shfl_xor(v, 2, 32));
      v = fmaxf(v, __shfl_xor(v, 4, 32));
      v = fmaxf(v, __shfl_xor(v, 8, 32));
      const float mnew  = fmaxf(mrow[r], v);
      const float alpha = __builtin_exp2f((mrow[r] - mnew) * SCL);
      mrow[r] = mnew;
      const float p0 = (s0[r] > -1.0e37f) ? __builtin_exp2f((s0[r] - mnew) * SCL) : 0.f;
      const float p1 = (s1[r] > -1.0e37f) ? __builtin_exp2f((s1[r] - mnew) * SCL) : 0.f;
      float ps = p0 + p1;
      ps += __shfl_xor(ps, 1, 32);
      ps += __shfl_xor(ps, 2, 32);
      ps += __shfl_xor(ps, 4, 32);
      ps += __shfl_xor(ps, 8, 32);
      lrow[r] = lrow[r] * alpha + ps;
      o[0][r] *= alpha; o[1][r] *= alpha; o[2][r] *= alpha; o[3][r] *= alpha;
      plds[wave][8 * hl + r][nloc]      = f2bf(p0);
      plds[wave][8 * hl + r][16 + nloc] = f2bf(p1);
    }
    __syncthreads();

    // ---- P back in A-operand layout
    const v16bf_t pa = lds_a_tile(&plds[wave][0][0], 40);
    __syncthreads();

    // ---- O += P * V : batch 4 LDS B-tiles, then 4 WMMAs
    v16bf_t vt4[4];
#pragma unroll
    for (int j = 0; j < 4; ++j) vt4[j] = lds_b_tile(&vbuf[cur][0], j * 16, 32, 0);
#pragma unroll
    for (int j = 0; j < 4; ++j) o[j] = wmma_bf16(pa, vt4[j], o[j]);

#if USE_TDM
    if (wave == 0) TDM_WAIT();
#endif
    __syncthreads();
  }

  // ---- normalize and store context (bf16)
#pragma unroll
  for (int r = 0; r < 8; ++r) {
    const float inv = (lrow[r] > 0.f) ? (1.f / lrow[r]) : 0.f;
#pragma unroll
    for (int j = 0; j < 4; ++j) {
      Ctx[aOff + (size_t)(qb + 8 * hl + r) * D_MODEL + dh0 + j * 16 + nloc] =
          f2bf(o[j][r] * inv);
    }
  }
}

// --------------------------------------------------------------------------
extern "C" void kernel_launch(void* const* d_in, const int* in_sizes, int n_in,
                              void* d_out, int out_size, void* d_ws, size_t ws_size,
                              hipStream_t stream) {
  const float* q  = (const float*)d_in[0];
  const float* k  = (const float*)d_in[1];
  const float* v  = (const float*)d_in[2];
  const unsigned char* mask = (const unsigned char*)d_in[3];
  const float* Wq = (const float*)d_in[4];
  const float* bq = (const float*)d_in[5];
  const float* Wk = (const float*)d_in[6];
  const float* bk = (const float*)d_in[7];
  const float* Wv = (const float*)d_in[8];
  const float* bv = (const float*)d_in[9];
  const float* Wo = (const float*)d_in[10];
  const float* bo = (const float*)d_in[11];

  const size_t SZ_ACT = (size_t)BB * SS * D_MODEL * 2;      // 8 MB
  const size_t SZ_W   = (size_t)D_MODEL * D_MODEL * 2;      // 2 MB
  char* ws = (char*)d_ws;
  unsigned short* Xq  = (unsigned short*)ws; ws += SZ_ACT;
  unsigned short* Xk  = (unsigned short*)ws; ws += SZ_ACT;
  unsigned short* Xv  = (unsigned short*)ws; ws += SZ_ACT;
  unsigned short* Wqb = (unsigned short*)ws; ws += SZ_W;
  unsigned short* Wkb = (unsigned short*)ws; ws += SZ_W;
  unsigned short* Wvb = (unsigned short*)ws; ws += SZ_W;
  unsigned short* Wob = (unsigned short*)ws; ws += SZ_W;
  unsigned short* Qp  = (unsigned short*)ws; ws += SZ_ACT;
  unsigned short* Kp  = (unsigned short*)ws; ws += SZ_ACT;
  unsigned short* Vtp = (unsigned short*)ws; ws += SZ_ACT;  // (B, D, S)
  unsigned short* Ctx = (unsigned short*)ws; ws += SZ_ACT;

  const int ACT_N4 = BB * SS * D_MODEL / 4;
  const int W_N4   = D_MODEL * D_MODEL / 4;
  cvt_kernel<<<ACT_N4 / 256, 256, 0, stream>>>(q,  Xq,  ACT_N4);
  cvt_kernel<<<ACT_N4 / 256, 256, 0, stream>>>(k,  Xk,  ACT_N4);
  cvt_kernel<<<ACT_N4 / 256, 256, 0, stream>>>(v,  Xv,  ACT_N4);
  cvt_kernel<<<W_N4   / 256, 256, 0, stream>>>(Wq, Wqb, W_N4);
  cvt_kernel<<<W_N4   / 256, 256, 0, stream>>>(Wk, Wkb, W_N4);
  cvt_kernel<<<W_N4   / 256, 256, 0, stream>>>(Wv, Wvb, W_N4);
  cvt_kernel<<<W_N4   / 256, 256, 0, stream>>>(Wo, Wob, W_N4);

  const dim3 gGemm(D_MODEL / 128, (BB * SS) / 128);         // (8, 32)
  gemm_bf16_kernel<<<gGemm, 256, 0, stream>>>(Xq, Wqb, bq, Qp, nullptr, nullptr, D_MODEL, D_MODEL);
  gemm_bf16_kernel<<<gGemm, 256, 0, stream>>>(Xk, Wkb, bk, Kp, nullptr, nullptr, D_MODEL, D_MODEL);
  gemm_bf16_kernel<<<gGemm, 256, 0, stream>>>(Xv, Wvb, bv, nullptr, nullptr, Vtp, D_MODEL, D_MODEL);

  attn_kernel<<<dim3(SS / 64, N_HEADS, BB), 128, 0, stream>>>(Qp, Kp, Vtp, mask, Ctx);

  gemm_bf16_kernel<<<gGemm, 256, 0, stream>>>(Ctx, Wob, bo, nullptr, (float*)d_out, nullptr,
                                              D_MODEL, D_MODEL);
}
